// PILayer_70815420776685
// MI455X (gfx1250) — matrix operand order repaired and checked by reading, not compiled, not run on hardware
//
#include <hip/hip_runtime.h>

// CDNA5 / gfx1250 PILayer kernel, v2:
//   out[e,:] = ( [prop[i],prop[j]] @ W1 + b1 ) -> h[e,64]
//   fused:  out = h'(e, k*8+b) @ W2r,  h' = h[e,k]*basis[e,b], K extended by
//           one 32-step carrying basis itself against b2 (bias*basis term).
// All matrix math via v_wmma_f32_16x16x32_bf16 (f32 accumulate).

typedef __attribute__((ext_vector_type(16))) __bf16 v16bf;
typedef __attribute__((ext_vector_type(8)))  float  v8f;
typedef __attribute__((ext_vector_type(2)))  __bf16 v2bf;

union FragB {
    v16bf v;
    uint4 q[2];
    unsigned int w[8];
};

// pack two fp32 into one dword of 2x bf16 (compiler picks native cvt if present)
__device__ __forceinline__ unsigned int pkbf(float lo, float hi) {
    v2bf r;
    r[0] = (__bf16)lo;
    r[1] = (__bf16)hi;
    return __builtin_bit_cast(unsigned int, r);
}

#define TILES_PER_BLOCK 8   // 8 waves of 32 = 256 threads
#define HSTRIDE 65          // padded f32 row stride (bank-conflict-free scalar reads)

__launch_bounds__(256, 1)
__global__ void pilayer_wmma_kernel(const float* __restrict__ prop,
                                    const int*   __restrict__ idx_i,
                                    const int*   __restrict__ idx_j,
                                    const float* __restrict__ basis,
                                    const float* __restrict__ W1,   // [128,64]
                                    const float* __restrict__ b1,   // [64]
                                    const float* __restrict__ W2,   // [64,512]
                                    const float* __restrict__ b2,   // [512]
                                    float* __restrict__ out,        // [E,64]
                                    int nEdges, int nTiles)
{
    // B-fragment layout (per 512-short fragment): shorts [0,256) = all lanes' low
    // 16B (lane*8 + e), shorts [256,512) = all lanes' high 16B. Both halves are
    // dense 512B wave reads -> conflict-free ds_load_b128.
    __shared__ unsigned short W1F[16 * 512];            // 16 KB : frags (q=0..3, n=0..3)
    __shared__ unsigned short W2F[68 * 512];            // 68 KB : frags (q=0..16, n3=0..3)
    __shared__ float HBf[TILES_PER_BLOCK][16 * HSTRIDE];// 32.5 KB : per-wave h (f32)

    const int tid  = threadIdx.x;
    const int lane = tid & 31;
    const int wave = tid >> 5;

    // ---- cooperative weight staging ----
    // B layout (32x16 bf16): lanes 0-15 K={0..7,16..23}, lanes 16-31 K={8..15,24..31}, N=lane%16,
    // element e => K = kbase + (e/8)*16 + (lane/16)*8 + e%8.
    for (int t = tid; t < 16 * 512; t += 256) {
        int fi = t >> 9, rem = t & 511;
        int half = rem >> 8, r2 = rem & 255;
        int l = r2 >> 3, e = half * 8 + (r2 & 7);
        int qq = fi >> 2, n = fi & 3;
        int K = qq * 32 + ((e >> 3) << 4) + ((l >> 4) << 3) + (e & 7);
        int N = n * 16 + (l & 15);
        unsigned int pk = pkbf(W1[K * 64 + N], 0.0f);
        W1F[t] = (unsigned short)(pk & 0xFFFFu);
    }
    // W2r[k*8+b, c] = W2[k, c*8+b]; rows 512..519 carry b2; 520..543 zero.
    for (int t = tid; t < 68 * 512; t += 256) {
        int fi = t >> 9, rem = t & 511;
        int half = rem >> 8, r2 = rem & 255;
        int l = r2 >> 3, e = half * 8 + (r2 & 7);
        int qq = fi >> 2, n3 = fi & 3;
        int K = qq * 32 + ((e >> 3) << 4) + ((l >> 4) << 3) + (e & 7);
        int c = n3 * 16 + (l & 15);
        float val;
        if (K < 512)       val = W2[(K >> 3) * 512 + c * 8 + (K & 7)];
        else if (K < 520)  val = b2[c * 8 + (K - 512)];
        else               val = 0.0f;
        unsigned int pk = pkbf(val, 0.0f);
        W2F[t] = (unsigned short)(pk & 0xFFFFu);
    }
    __syncthreads();

    const int halfHi = lane >> 4;   // 0: lanes 0-15, 1: lanes 16-31
    const int lcol   = lane & 15;

    for (int tile = blockIdx.x * TILES_PER_BLOCK + wave; tile < nTiles;
         tile += gridDim.x * TILES_PER_BLOCK) {

        const int e0 = tile * 16;
        int eLane = e0 + lcol;
        if (eLane >= nEdges) eLane = nEdges - 1;   // clamp (E is a multiple of 16 anyway)

        const float* rowi = prop + (long)idx_i[eLane] * 64;
        const float* rowj = prop + (long)idx_j[eLane] * 64;

        // full basis row for this lane's A-row edge
        float4 bA = *(const float4*)(basis + (long)eLane * 8);
        float4 bB = *(const float4*)(basis + (long)eLane * 8 + 4);
        const float basf[8] = { bA.x, bA.y, bA.z, bA.w, bB.x, bB.y, bB.z, bB.w };
        const unsigned int pb0 = pkbf(basf[0], basf[1]);
        const unsigned int pb1 = pkbf(basf[2], basf[3]);
        const unsigned int pb2w = pkbf(basf[4], basf[5]);
        const unsigned int pb3 = pkbf(basf[6], basf[7]);

        // ---- stage-1 A fragments (16x32 bf16, A layout) from gathered prop rows ----
        // lane<16 holds K = kbase + {0..7,16..23}; lane>=16: kbase + {8..15,24..31}.
        FragB afr[4];
#pragma unroll
        for (int qq = 0; qq < 4; ++qq) {
            const float* row = (qq < 2) ? rowi : rowj;
            const int f0 = (qq & 1) * 32 + halfHi * 8;
            float4 c0 = *(const float4*)(row + f0);
            float4 c1 = *(const float4*)(row + f0 + 4);
            float4 c2 = *(const float4*)(row + f0 + 16);
            float4 c3 = *(const float4*)(row + f0 + 20);
            afr[qq].w[0] = pkbf(c0.x, c0.y);
            afr[qq].w[1] = pkbf(c0.z, c0.w);
            afr[qq].w[2] = pkbf(c1.x, c1.y);
            afr[qq].w[3] = pkbf(c1.z, c1.w);
            afr[qq].w[4] = pkbf(c2.x, c2.y);
            afr[qq].w[5] = pkbf(c2.z, c2.w);
            afr[qq].w[6] = pkbf(c3.x, c3.y);
            afr[qq].w[7] = pkbf(c3.z, c3.w);
        }

        // ---- stage 1: h[16,64] = A(16x128) @ W1 + b1, store f32 to HBf ----
#pragma unroll
        for (int n = 0; n < 4; ++n) {
            float b1v = b1[n * 16 + lcol];
            v8f acc;
#pragma unroll
            for (int r = 0; r < 8; ++r) acc[r] = b1v;
#pragma unroll
            for (int qq = 0; qq < 4; ++qq) {
                FragB bfr;
                const int fb = (qq * 4 + n) * 512;
                bfr.q[0] = *(const uint4*)&W1F[fb + lane * 8];
                bfr.q[1] = *(const uint4*)&W1F[fb + 256 + lane * 8];
                acc = __builtin_amdgcn_wmma_f32_16x16x32_bf16(
                          false, afr[qq].v, false, bfr.v, (short)0, acc, false, false);
            }
            // D layout: lane holds col N = n*16+lcol, rows M = r + halfHi*8
#pragma unroll
            for (int r = 0; r < 8; ++r)
                HBf[wave][(r + halfHi * 8) * HSTRIDE + (n * 16 + lcol)] = acc[r];
        }

        // wave-internal cross-lane transpose through LDS
        asm volatile("s_wait_dscnt 0" ::: "memory");

        // ---- stage 2: out[16,64] = h'(16x544) @ W2r(544x64) ----
        const float* hrow = &HBf[wave][lcol * HSTRIDE];
        v8f acc2[4];
#pragma unroll
        for (int n3 = 0; n3 < 4; ++n3)
#pragma unroll
            for (int r = 0; r < 8; ++r) acc2[n3][r] = 0.0f;

#pragma unroll 4
        for (int qq = 0; qq < 16; ++qq) {
            // A chunk (8 consecutive K) = h[e, k] * basis[e, 0..7]
            const float h0 = hrow[qq * 4 + halfHi];
            const float h1 = hrow[qq * 4 + halfHi + 2];
            FragB a2;
            a2.w[0] = pkbf(h0 * basf[0], h0 * basf[1]);
            a2.w[1] = pkbf(h0 * basf[2], h0 * basf[3]);
            a2.w[2] = pkbf(h0 * basf[4], h0 * basf[5]);
            a2.w[3] = pkbf(h0 * basf[6], h0 * basf[7]);
            a2.w[4] = pkbf(h1 * basf[0], h1 * basf[1]);
            a2.w[5] = pkbf(h1 * basf[2], h1 * basf[3]);
            a2.w[6] = pkbf(h1 * basf[4], h1 * basf[5]);
            a2.w[7] = pkbf(h1 * basf[6], h1 * basf[7]);
#pragma unroll
            for (int n3 = 0; n3 < 4; ++n3) {
                FragB bfr;
                const int fb = (qq * 4 + n3) * 512;
                bfr.q[0] = *(const uint4*)&W2F[fb + lane * 8];
                bfr.q[1] = *(const uint4*)&W2F[fb + 256 + lane * 8];
                acc2[n3] = __builtin_amdgcn_wmma_f32_16x16x32_bf16(
                               false, a2.v, false, bfr.v, (short)0, acc2[n3], false, false);
            }
        }
        { // extension K-step: A rows carry basis (against b2 rows of W2r)
            FragB a2;
            a2.w[0] = halfHi ? 0u : pb0;
            a2.w[1] = halfHi ? 0u : pb1;
            a2.w[2] = halfHi ? 0u : pb2w;
            a2.w[3] = halfHi ? 0u : pb3;
            a2.w[4] = 0u; a2.w[5] = 0u; a2.w[6] = 0u; a2.w[7] = 0u;
#pragma unroll
            for (int n3 = 0; n3 < 4; ++n3) {
                FragB bfr;
                const int fb = (64 + n3) * 512;
                bfr.q[0] = *(const uint4*)&W2F[fb + lane * 8];
                bfr.q[1] = *(const uint4*)&W2F[fb + 256 + lane * 8];
                acc2[n3] = __builtin_amdgcn_wmma_f32_16x16x32_bf16(
                               false, a2.v, false, bfr.v, (short)0, acc2[n3], false, false);
            }
        }

        // ---- store: D layout -> out[e, c] ----
#pragma unroll
        for (int n3 = 0; n3 < 4; ++n3) {
#pragma unroll
            for (int r = 0; r < 8; ++r) {
                const int edge = e0 + r + halfHi * 8;
                if (edge < nEdges)
                    out[(long)edge * 64 + n3 * 16 + lcol] = acc2[n3][r];
            }
        }
    }
}

extern "C" void kernel_launch(void* const* d_in, const int* in_sizes, int n_in,
                              void* d_out, int out_size, void* d_ws, size_t ws_size,
                              hipStream_t stream) {
    const float* prop  = (const float*)d_in[0];
    const int*   idx_i = (const int*)  d_in[1];
    const int*   idx_j = (const int*)  d_in[2];
    const float* basis = (const float*)d_in[3];
    const float* W1    = (const float*)d_in[4];
    const float* b1    = (const float*)d_in[5];
    const float* W2    = (const float*)d_in[6];
    const float* b2    = (const float*)d_in[7];
    float* out = (float*)d_out;

    const int nEdges = in_sizes[1];
    const int nTiles = (nEdges + 15) / 16;

    int blocks = (nTiles + TILES_PER_BLOCK - 1) / TILES_PER_BLOCK;
    if (blocks > 1024) blocks = 1024;

    pilayer_wmma_kernel<<<blocks, 256, 0, stream>>>(
        prop, idx_i, idx_j, basis, W1, b1, W2, b2, out, nEdges, nTiles);
}
